// SeqSeqModel_42958262894769
// MI455X (gfx1250) — matrix operand order repaired and checked by reading, not compiled, not run on hardware
//
#include <hip/hip_runtime.h>

// Problem constants (match reference)
#define HB   8      // batch
#define TT   128    // timesteps
#define NN   128    // nodes
#define HH   256    // hidden
#define KIN  16     // input features
#define EE   4096   // edges
#define LN_EPSF 1e-5f
#define MIW  288    // padded mi width (256 agg + 16 x + 16 zero), multiple of 32

typedef __attribute__((ext_vector_type(16))) _Float16 v16h;
typedef __attribute__((ext_vector_type(8)))  _Float16 v8h;
typedef __attribute__((ext_vector_type(8)))  float    v8f;

__device__ __forceinline__ int lane_id() { return (int)(threadIdx.x & 31u); }

// ---------------- WMMA fragment helpers (wave32, 16x16x32 f16) ----------------
// A fragment: 16x32 (MxK), row-major source with leading dim ld (halfs).
// lane L: row = L%16 ; a[0..7] = K 8*(L/16)+0..7 ; a[8..15] = K 16+8*(L/16)+0..7
__device__ __forceinline__ v16h frag_a(const _Float16* p0, int ld) {
  int l = lane_id();
  const _Float16* p = p0 + (l & 15) * ld + ((l >> 4) << 3);
  v16h r;
  *(v8h*)(&r)       = *(const v8h*)(p);
  *((v8h*)(&r) + 1) = *(const v8h*)(p + 16);
  return r;
}
// B fragment: 32x16 (KxN). Source is row-major W[n][k] (i.e. B = W^T), ld in halfs.
// lane L: col n = L%16 ; b[0..7] = K 8*(L/16)+0..7 ; b[8..15] = K 16+8*(L/16)+0..7
__device__ __forceinline__ v16h frag_b(const _Float16* p0, int ld) {
  int l = lane_id();
  const _Float16* p = p0 + (l & 15) * ld + ((l >> 4) << 3);
  v16h r;
  *(v8h*)(&r)       = *(const v8h*)(p);
  *((v8h*)(&r) + 1) = *(const v8h*)(p + 16);
  return r;
}
__device__ __forceinline__ v8f wmma_f16(v16h a, v16h b, v8f c) {
  return __builtin_amdgcn_wmma_f32_16x16x32_f16(false, a, false, b, (short)0, c,
                                                false, false);
}

// Generic tiled GEMM: Out[M x N] = A[M x K] * W^T (+ colBias) (+ elemBias), fp16 out.
// A: LDS row-major (lda). W: row-major [N][K] (ldb). Optional transposed store.
template <bool TRANS_OUT>
__device__ void gemm_phase(const _Float16* A, int lda,
                           const _Float16* Bw, int ldb,
                           const float* colBias,
                           const float* elemBias, int ldeb,
                           _Float16* Out, int ldo,
                           int Mtiles, int Ntiles, int Ktiles,
                           int wave, int nwaves) {
  int l = lane_id();
  int col = l & 15;
  int r0  = (l >> 4) * 8;
  for (int t = wave; t < Mtiles * Ntiles; t += nwaves) {
    int m0 = (t % Mtiles) * 16;
    int n0 = (t / Mtiles) * 16;
    v8f acc = {};
    for (int k = 0; k < Ktiles; ++k) {
      v16h a = frag_a(A + m0 * lda + 32 * k, lda);
      v16h b = frag_b(Bw + n0 * ldb + 32 * k, ldb);
      acc = wmma_f16(a, b, acc);
    }
    float cb = colBias ? colBias[n0 + col] : 0.0f;
#pragma unroll
    for (int i = 0; i < 8; ++i) {
      int row = m0 + r0 + i;
      float v = acc[i] + cb;
      if (elemBias) v += elemBias[row * ldeb + n0 + col];
      if (TRANS_OUT) Out[(n0 + col) * ldo + row] = (_Float16)v;
      else           Out[row * ldo + (n0 + col)] = (_Float16)v;
    }
  }
}

__device__ __forceinline__ float wave_sum(float v) {
#pragma unroll
  for (int off = 16; off > 0; off >>= 1) v += __shfl_xor(v, off, 32);
  return v;
}
__device__ __forceinline__ float sigmoidf(float x) {
  return 1.0f / (1.0f + __expf(-x));
}

// ---------------- main persistent per-batch kernel ----------------
__global__ __launch_bounds__(512, 1)
void gru_graph_kernel(const float* __restrict__ x,
                      const float* __restrict__ targets,
                      const unsigned char* __restrict__ tmask,
                      const float* __restrict__ b_s2m,
                      const float* __restrict__ b_mix,
                      const float* __restrict__ ln_g,
                      const float* __restrict__ ln_b,
                      const float* __restrict__ b_ih,
                      const float* __restrict__ b_hh,
                      const float* __restrict__ W_out,
                      const float* __restrict__ b_out,
                      const _Float16* __restrict__ Ws2m16,
                      const _Float16* __restrict__ Wmix16,
                      const _Float16* __restrict__ Wih16,
                      const _Float16* __restrict__ Whh16,
                      const _Float16* __restrict__ Adj16,
                      const float* __restrict__ edge_bias,
                      float* __restrict__ Sg0base,   // fp32 state master ping
                      float* __restrict__ Sg1base,   // fp32 state master pong
                      float* __restrict__ loss_out,
                      float* __restrict__ outs_out) {
  // LDS: 2*64KB fp16 state ping-pong + 64KB m^T/m2 + 72KB mi + 32KB Adj = 296KB
  __shared__ _Float16 S0[NN * HH];
  __shared__ _Float16 S1[NN * HH];
  __shared__ _Float16 MB[HH * NN];    // phase1: m^T [HH][NN]; phase3/4: m2 [NN][HH]
  __shared__ _Float16 MI[NN * MIW];   // [agg | xt | 0]
  __shared__ _Float16 ADJ[NN * NN];

  const int b    = blockIdx.x;
  const int tid  = threadIdx.x;
  const int wave = tid >> 5;
  const int NW   = (int)(blockDim.x >> 5);
  const int lane = lane_id();

  float* Sg0 = Sg0base + (size_t)b * NN * HH;  // fp32 recurrence carry
  float* Sg1 = Sg1base + (size_t)b * NN * HH;

  // init: zero fp16 + fp32 state, stage Adj into LDS
  for (int i = tid; i < NN * HH; i += blockDim.x) {
    S0[i] = (_Float16)0.0f;
    Sg0[i] = 0.0f;
  }
  for (int i = tid; i < NN * NN; i += blockDim.x) ADJ[i] = Adj16[i];
  __syncthreads();

#pragma unroll 1
  for (int t = 0; t < TT; ++t) {
    _Float16* Sold = (t & 1) ? S1 : S0;
    _Float16* Snew = (t & 1) ? S0 : S1;
    float*    SoldG = (t & 1) ? Sg1 : Sg0;
    float*    SnewG = (t & 1) ? Sg0 : Sg1;

    // P1: m = Sold @ Ws2m^T + b_s2m, stored transposed -> MB as m^T [HH][NN]
    gemm_phase<true>(Sold, HH, Ws2m16, HH, b_s2m, nullptr, 0,
                     MB, NN, NN / 16, HH / 16, HH / 32, wave, NW);
    // prefetch next timestep's x slice (8KB = 64 cachelines) behind the GEMM
    if (t + 1 < TT && tid < 64) {
      const float* nx = x + (size_t)(b * TT + t + 1) * NN * KIN + tid * 32;
      __builtin_prefetch(nx, 0, 3);
    }
    __syncthreads();

    // P2: agg = Adj @ m + edge_bias  -> MI[:,0:256]; B-side = m^T (row-major [h][src])
    gemm_phase<false>(ADJ, NN, MB, NN, nullptr, edge_bias, HH,
                      MI, MIW, NN / 16, HH / 16, NN / 32, wave, NW);
    // fill MI[:,256:288] with xt (fp16) and zero pad
    for (int i = tid; i < NN * 32; i += blockDim.x) {
      int node = i >> 5, c = i & 31;
      float v = (c < KIN) ? x[((size_t)(b * TT + t) * NN + node) * KIN + c] : 0.0f;
      MI[node * MIW + HH + c] = (_Float16)v;
    }
    __syncthreads();

    // P3: m2 = MI @ Wmix^T + b_mix -> MB reused as [NN][HH]
    gemm_phase<false>(MI, MIW, Wmix16, MIW, b_mix, nullptr, 0,
                      MB, HH, NN / 16, HH / 16, MIW / 32, wave, NW);
    __syncthreads();

    // P4: LayerNorm rows of MB in-place (one wave per row, 8 rows/wave)
    for (int row = wave; row < NN; row += NW) {
      float vals[8];
      float s = 0.0f, ss = 0.0f;
#pragma unroll
      for (int j = 0; j < 8; ++j) {
        float v = (float)MB[row * HH + lane + 32 * j];
        vals[j] = v; s += v; ss += v * v;
      }
      s  = wave_sum(s);
      ss = wave_sum(ss);
      float mu  = s * (1.0f / HH);
      float var = ss * (1.0f / HH) - mu * mu;
      float inv = rsqrtf(var + LN_EPSF);
#pragma unroll
      for (int j = 0; j < 8; ++j) {
        int h = lane + 32 * j;
        MB[row * HH + h] = (_Float16)((vals[j] - mu) * inv * ln_g[h] + ln_b[h]);
      }
    }
    __syncthreads();

    // P5: GRU gates. gi = m2 @ Wih^T, gh = Sold @ Whh^T, fused per 16x16 tile.
    //     Recurrence carry in fp32 (global master); fp16 copy feeds next WMMAs.
    {
      int col = lane & 15;
      int r0  = (lane >> 4) * 8;
      for (int pos = wave; pos < (NN / 16) * (HH / 16); pos += NW) {
        int m0 = (pos % (NN / 16)) * 16;
        int h0 = (pos / (NN / 16)) * 16;
        v8f air = {}, aiz = {}, ain = {}, ahr = {}, ahz = {}, ahn = {};
#pragma unroll
        for (int k = 0; k < HH / 32; ++k) {
          v16h am = frag_a(MB   + m0 * HH + 32 * k, HH);
          v16h as = frag_a(Sold + m0 * HH + 32 * k, HH);
          v16h br = frag_b(Wih16 + (size_t)(0 * HH + h0) * HH + 32 * k, HH);
          v16h bz = frag_b(Wih16 + (size_t)(1 * HH + h0) * HH + 32 * k, HH);
          v16h bn = frag_b(Wih16 + (size_t)(2 * HH + h0) * HH + 32 * k, HH);
          v16h cr = frag_b(Whh16 + (size_t)(0 * HH + h0) * HH + 32 * k, HH);
          v16h cz = frag_b(Whh16 + (size_t)(1 * HH + h0) * HH + 32 * k, HH);
          v16h cn = frag_b(Whh16 + (size_t)(2 * HH + h0) * HH + 32 * k, HH);
          air = wmma_f16(am, br, air);
          aiz = wmma_f16(am, bz, aiz);
          ain = wmma_f16(am, bn, ain);
          ahr = wmma_f16(as, cr, ahr);
          ahz = wmma_f16(as, cz, ahz);
          ahn = wmma_f16(as, cn, ahn);
        }
        int h = h0 + col;
        float bir = b_ih[h], biz = b_ih[HH + h], bin = b_ih[2 * HH + h];
        float bhr = b_hh[h], bhz = b_hh[HH + h], bhn = b_hh[2 * HH + h];
#pragma unroll
        for (int i = 0; i < 8; ++i) {
          int row = m0 + r0 + i;
          float sold = SoldG[row * HH + h];  // fp32 carry
          float r = sigmoidf(air[i] + bir + ahr[i] + bhr);
          float z = sigmoidf(aiz[i] + biz + ahz[i] + bhz);
          float n = tanhf(ain[i] + bin + r * (ahn[i] + bhn));
          float hnew = (1.0f - z) * n + z * sold;
          SnewG[row * HH + h] = hnew;          // fp32 master
          Snew[row * HH + h]  = (_Float16)hnew; // fp16 for next WMMAs
        }
      }
    }
    __syncthreads();

    // P6: out = SnewG @ W_out^T + b_out ; loss = mask * (out - target)^2
    for (int node = wave; node < NN; node += NW) {
      float acc = 0.0f;
#pragma unroll
      for (int j = 0; j < 8; ++j) {
        int h = lane + 32 * j;
        acc += SnewG[node * HH + h] * W_out[h];
      }
      acc = wave_sum(acc);
      if (lane == 0) {
        float o = acc + b_out[0];
        size_t idx = (size_t)(b * TT + t) * NN + node;
        outs_out[idx] = o;
        float d = o - targets[idx];
        loss_out[idx] = tmask[idx] ? d * d : 0.0f;
      }
    }
    __syncthreads();
  }
}

// ---------------- prep kernels (run once per launch, deterministic) ----------------
__global__ void k_zero_adj(int* adjCnt) {
  int i = blockIdx.x * blockDim.x + threadIdx.x;
  if (i < NN * NN) adjCnt[i] = 0;
}
__global__ void k_count_adj(const int* __restrict__ graph, int* adjCnt) {
  int e = blockIdx.x * blockDim.x + threadIdx.x;
  if (e < EE) {
    int s = graph[e];
    int t = graph[EE + e];
    atomicAdd(&adjCnt[t * NN + s], 1);
  }
}
// deterministic edge_bias: per (node,h) scan over edges (one-time cost)
__global__ void k_edge_bias(const int* __restrict__ graph,
                            const float* __restrict__ edge_emb,
                            float* __restrict__ eb) {
  int i = blockIdx.x * blockDim.x + threadIdx.x;
  if (i < NN * HH) {
    int n = i / HH, h = i % HH;
    const int* tg = graph + EE;
    float a = 0.0f;
    for (int e = 0; e < EE; ++e)
      if (tg[e] == n) a += edge_emb[(size_t)e * HH + h];
    eb[i] = a;
  }
}
__global__ void k_cvt(const float* __restrict__ s, _Float16* __restrict__ d, int n) {
  int i = blockIdx.x * blockDim.x + threadIdx.x;
  if (i < n) d[i] = (_Float16)s[i];
}
__global__ void k_cvt_adj(const int* __restrict__ c, _Float16* __restrict__ d) {
  int i = blockIdx.x * blockDim.x + threadIdx.x;
  if (i < NN * NN) d[i] = (_Float16)(float)c[i];
}
__global__ void k_cvt_wmix(const float* __restrict__ s, _Float16* __restrict__ d) {
  int i = blockIdx.x * blockDim.x + threadIdx.x;
  if (i < HH * MIW) {
    int o = i / MIW, k = i % MIW;
    d[i] = (k < HH + KIN) ? (_Float16)s[o * (HH + KIN) + k] : (_Float16)0.0f;
  }
}

extern "C" void kernel_launch(void* const* d_in, const int* in_sizes, int n_in,
                              void* d_out, int out_size, void* d_ws, size_t ws_size,
                              hipStream_t stream) {
  const float* x        = (const float*)d_in[0];
  const float* targets  = (const float*)d_in[1];
  const unsigned char* tmask = (const unsigned char*)d_in[2];
  const int*   graph    = (const int*)d_in[3];
  const float* W_s2m    = (const float*)d_in[4];
  const float* b_s2m    = (const float*)d_in[5];
  const float* edge_emb = (const float*)d_in[6];
  const float* W_mix    = (const float*)d_in[7];
  const float* b_mix    = (const float*)d_in[8];
  const float* ln_g     = (const float*)d_in[9];
  const float* ln_b     = (const float*)d_in[10];
  const float* W_ih     = (const float*)d_in[11];
  const float* W_hh     = (const float*)d_in[12];
  const float* b_ih     = (const float*)d_in[13];
  const float* b_hh     = (const float*)d_in[14];
  const float* W_out    = (const float*)d_in[15];
  const float* b_out    = (const float*)d_in[16];

  char* ws = (char*)d_ws;
  size_t off = 0;
  auto take = [&](size_t bytes) -> void* {
    void* p = ws + off;
    off = (off + bytes + 255) & ~(size_t)255;
    return p;
  };
  int*       adjCnt    = (int*)take(NN * NN * sizeof(int));
  _Float16*  Adj16     = (_Float16*)take(NN * NN * sizeof(_Float16));
  float*     edge_bias = (float*)take(NN * HH * sizeof(float));
  _Float16*  Ws2m16    = (_Float16*)take((size_t)HH * HH * sizeof(_Float16));
  _Float16*  Wmix16    = (_Float16*)take((size_t)HH * MIW * sizeof(_Float16));
  _Float16*  Wih16     = (_Float16*)take((size_t)3 * HH * HH * sizeof(_Float16));
  _Float16*  Whh16     = (_Float16*)take((size_t)3 * HH * HH * sizeof(_Float16));
  float*     Sg0       = (float*)take((size_t)HB * NN * HH * sizeof(float));
  float*     Sg1       = (float*)take((size_t)HB * NN * HH * sizeof(float));

  k_zero_adj<<<(NN * NN + 255) / 256, 256, 0, stream>>>(adjCnt);
  k_count_adj<<<(EE + 255) / 256, 256, 0, stream>>>(graph, adjCnt);
  k_edge_bias<<<(NN * HH + 255) / 256, 256, 0, stream>>>(graph, edge_emb, edge_bias);
  k_cvt<<<(HH * HH + 255) / 256, 256, 0, stream>>>(W_s2m, Ws2m16, HH * HH);
  k_cvt<<<(3 * HH * HH + 255) / 256, 256, 0, stream>>>(W_ih, Wih16, 3 * HH * HH);
  k_cvt<<<(3 * HH * HH + 255) / 256, 256, 0, stream>>>(W_hh, Whh16, 3 * HH * HH);
  k_cvt_wmix<<<(HH * MIW + 255) / 256, 256, 0, stream>>>(W_mix, Wmix16);
  k_cvt_adj<<<(NN * NN + 255) / 256, 256, 0, stream>>>(adjCnt, Adj16);

  float* loss_out = (float*)d_out;
  float* outs_out = (float*)d_out + (size_t)HB * TT * NN;

  gru_graph_kernel<<<HB, 512, 0, stream>>>(
      x, targets, tmask, b_s2m, b_mix, ln_g, ln_b, b_ih, b_hh, W_out, b_out,
      Ws2m16, Wmix16, Wih16, Whh16, Adj16, edge_bias, Sg0, Sg1,
      loss_out, outs_out);
}
